// ScaledDotProductAttention_60344290509508
// MI455X (gfx1250) — compile-verified, matrix operand order checked by reading, not run on hardware
//
#include <hip/hip_runtime.h>
#include <hip/hip_bf16.h>

// ---------------------------------------------------------------------------
// Fused causal attention for MI455X (gfx1250, wave32, WMMA, TDM, async-LDS).
//   B=8, S=2048, D=1024.  All matmuls bf16 WMMA (v_wmma_f32_16x16x32_bf16),
//   f32 accumulation, f32 softmax stats.  k/q tiles staged by the Tensor Data
//   Mover (tensor_load_to_lds, TENSORcnt), P tiles by global_load_async_to_lds
//   (ASYNCcnt), double-buffered against WMMA compute.
// ---------------------------------------------------------------------------

typedef __bf16 bf16;
typedef __attribute__((ext_vector_type(16))) __bf16 v16bf;
typedef __attribute__((ext_vector_type(8)))  float  v8f;
typedef __attribute__((ext_vector_type(4)))  unsigned tdm_v4u;
typedef __attribute__((ext_vector_type(8)))  int      tdm_v8i;
typedef __attribute__((ext_vector_type(4)))  int      tdm_v4i;

constexpr int Bn = 8;
constexpr int Sn = 2048;
constexpr int Dn = 1024;
constexpr float SCALE = 0.03125f;   // 1/sqrt(1024)

union FragU { v16bf v; unsigned u[8]; };

// 16x32 bf16 WMMA A/B^T fragment from an LDS row-major row (K contiguous).
// ISA 7.12.2: lanes 0-15 hold K {0..7,16..23}; lanes 16-31 hold K {8..15,
// 24..31}; each VGPR packs a consecutive K pair -> lowers to 2x ds_load_b128.
__device__ inline v16bf frag_ld(const unsigned* row32, int half) {
  FragU f;
#pragma unroll
  for (int j = 0; j < 4; ++j) f.u[j] = row32[half * 4 + j];
#pragma unroll
  for (int j = 0; j < 4; ++j) f.u[4 + j] = row32[8 + half * 4 + j];
  return f.v;
}

__device__ inline v8f wmma_bf16(v16bf a, v16bf b, v8f c) {
  return __builtin_amdgcn_wmma_f32_16x16x32_bf16(false, a, false, b, (short)0, c,
                                                 false, false);
}

__device__ inline unsigned rfl(unsigned x) {
  return (unsigned)__builtin_amdgcn_readfirstlane((int)x);
}

// ---------------------------------------------------------------------------
// Tensor Data Mover: 2D tile of bf16 (tileW x tileH elems, row stride in
// elems) from global -> LDS.  D# packed per ISA 8.3-8.6 (normal mode, no
// padding/iterate/gather, tensor dims set large so no OOB clamp).
// Tracked by TENSORcnt; pair with __builtin_amdgcn_s_wait_tensorcnt.
// This toolchain's builtin takes 6 args: (v4u g0, v8i g1, v4i g2, v4i g3,
// v8i pad/aux, i32 cpol) -- the aux group is passed zero-filled.
// ---------------------------------------------------------------------------
__device__ inline void tdm_load_2d(unsigned ldsOff, const void* gsrc,
                                   unsigned tileW, unsigned tileH,
                                   unsigned strideElems) {
  unsigned long long ga = (unsigned long long)gsrc;
  const unsigned TD = 1u << 20;              // huge logical tensor dims
  tdm_v4u g0;
  g0[0] = rfl(1u);                                        // count=1 (valid D#)
  g0[1] = rfl(ldsOff);                                    // lds_addr (bytes)
  g0[2] = rfl((unsigned)ga);                              // global_addr[31:0]
  g0[3] = rfl((unsigned)(ga >> 32) | (2u << 30));         // ga[56:32] | type=2
  tdm_v8i g1;
  g1[0] = (int)rfl(1u << 16);                             // data_size=2B
  g1[1] = (int)rfl((TD & 0xffffu) << 16);                 // tensor_dim0 lo16
  g1[2] = (int)rfl((TD >> 16) | ((TD & 0xffffu) << 16));  // td0 hi | td1 lo
  g1[3] = (int)rfl((TD >> 16) | (tileW << 16));           // td1 hi | tile_dim0
  g1[4] = (int)rfl(tileH);                                // tile_dim1 (dim2=0)
  g1[5] = (int)rfl(strideElems);                          // dim0_stride lo32
  g1[6] = 0;                                              // stride hi | d1s lo
  g1[7] = 0;                                              // d1s hi (unused)
  tdm_v4i g2;
  g2[0] = (int)TD; g2[1] = (int)TD; g2[2] = 0; g2[3] = 0; // dims 2/3, no tile3
  tdm_v4i g3 = {0, 0, 0, 0};
  tdm_v8i gx = {0, 0, 0, 0, 0, 0, 0, 0};
  __builtin_amdgcn_tensor_load_to_lds(g0, g1, g2, g3, gx, 0);
}

// Async global -> LDS 16B copy (GVS form), tracked by ASYNCcnt.
__device__ inline void async_ld_b128(unsigned ldsOff, const void* base, int voff) {
  asm volatile("global_load_async_to_lds_b128 %0, %1, %2"
               :: "v"(ldsOff), "v"(voff), "s"(base) : "memory");
}
__device__ inline void wait_async0() {
  asm volatile("s_wait_asynccnt 0x0" ::: "memory");
}

// ---------------------------------------------------------------------------
// Kernel 1: projection GEMM  Y[m,n] = bf16( sum_k X[m,k]*W[k,n] + bias[n] )
//   M = B*S = 16384, N = K = 1024.  Block tile 128x128, 256 threads (8 waves).
// ---------------------------------------------------------------------------
__global__ __launch_bounds__(256)
void proj_kernel(const float* __restrict__ X, const float* __restrict__ W,
                 const float* __restrict__ bias, bf16* __restrict__ Y) {
  __shared__ bf16 As[128 * 32];   // A tile, row-major [row][k]
  __shared__ bf16 Wt[128 * 32];   // W tile transposed [n][k] (B^T layout)

  const int tid  = threadIdx.x;
  const int lane = tid & 31;
  const int wave = tid >> 5;
  const int half = lane >> 4;
  const int l16  = lane & 15;
  const int n0   = blockIdx.x * 128;
  const int m0   = blockIdx.y * 128;

  v8f acc[8] = {};

  for (int k0 = 0; k0 < Dn; k0 += 32) {
    __syncthreads();
    // --- stage A: thread t -> row t/2, cols (t&1)*16..+15, f32 -> bf16 ---
    {
      int row = tid >> 1;
      int c0  = (tid & 1) * 16;
      const float* src = X + (size_t)(m0 + row) * Dn + k0 + c0;
#pragma unroll
      for (int j = 0; j < 16; j += 4) {
        float4 f4 = *(const float4*)(src + j);
        union { bf16 h[4]; uint2 u2; } p;
        p.h[0] = (bf16)f4.x; p.h[1] = (bf16)f4.y;
        p.h[2] = (bf16)f4.z; p.h[3] = (bf16)f4.w;
        *(uint2*)&As[row * 32 + c0 + j] = p.u2;
      }
    }
    // --- stage W transposed, K-pair packed (b32 LDS stores) ---
    // thread t: K pair (k0+2p, k0+2p+1) with p = t&15; n segment (t>>4)*8.
    {
      int p    = tid & 15;
      int nseg = tid >> 4;
      const float* r0 = W + (size_t)(k0 + 2 * p) * Dn + n0 + nseg * 8;
      const float* r1 = r0 + Dn;
      float4 a0 = ((const float4*)r0)[0], a1 = ((const float4*)r0)[1];
      float4 b0 = ((const float4*)r1)[0], b1 = ((const float4*)r1)[1];
      float fa[8] = {a0.x, a0.y, a0.z, a0.w, a1.x, a1.y, a1.z, a1.w};
      float fb[8] = {b0.x, b0.y, b0.z, b0.w, b1.x, b1.y, b1.z, b1.w};
#pragma unroll
      for (int j = 0; j < 8; ++j) {
        union { bf16 h[2]; unsigned u; } pk;
        pk.h[0] = (bf16)fa[j]; pk.h[1] = (bf16)fb[j];
        *(unsigned*)&Wt[(nseg * 8 + j) * 32 + 2 * p] = pk.u;
      }
    }
    __syncthreads();

    v16bf a = frag_ld((const unsigned*)&As[(wave * 16 + l16) * 32], half);
#pragma unroll
    for (int f = 0; f < 8; ++f) {
      v16bf b = frag_ld((const unsigned*)&Wt[(f * 16 + l16) * 32], half);
      acc[f] = wmma_bf16(a, b, acc[f]);
    }
  }

#pragma unroll
  for (int f = 0; f < 8; ++f) {
    int col = n0 + f * 16 + l16;
    float bv = bias[col];
#pragma unroll
    for (int r = 0; r < 8; ++r) {
      int row = m0 + wave * 16 + half * 8 + r;
      Y[(size_t)row * Dn + col] = (bf16)(acc[f][r] + bv);
    }
  }
}

// ---------------------------------------------------------------------------
// Shared score-tile machinery for kernels 2 & 3.
// Dynamic LDS map (offsets from 0):
//   [0, 128K)        q tile  64 x 1024 bf16   (resident, one TDM descriptor)
//   [128K, 144K)     k chunk buffer 0, 64 x 128 bf16
//   [144K, 160K)     k chunk buffer 1
// ---------------------------------------------------------------------------
constexpr unsigned QS_OFF  = 0;
constexpr unsigned KC_OFF  = 64 * Dn * 2;        // 131072
constexpr unsigned KC_BYTES = 64 * 128 * 2;      // 16384
constexpr int SMEM_SCORE = (int)(KC_OFF + 2 * KC_BYTES);   // 160 KB

// Computes the 64x64 score tile (qt,kt) into sc[4] for this wave, streaming
// k through double-buffered TDM chunks.  Caller must have issued the q-tile
// TDM and the chunk-0 TDM (wave 0) before entry; on exit TENSORcnt==0.
__device__ inline void score_tile(const bf16* qs_, const bf16* kc_,
                                  const bf16* ksrc, int wave, int l16, int half,
                                  v8f sc[4]) {
  for (int c = 0; c < 8; ++c) {
    __syncthreads();                 // previous chunk's consumers are done
    if (wave == 0) {
      if (c < 7) {
        tdm_load_2d(KC_OFF + (unsigned)((c + 1) & 1) * KC_BYTES,
                    ksrc + (c + 1) * 128, 128, 64, Dn);
        __builtin_amdgcn_s_wait_tensorcnt(1);   // chunk c (and q tile) landed
      } else {
        __builtin_amdgcn_s_wait_tensorcnt(0);
      }
    }
    __syncthreads();                 // chunk c visible to all waves
    const bf16* kcur = kc_ + (c & 1) * (KC_BYTES / 2);
    const int d0 = c * 128;
#pragma unroll
    for (int kk = 0; kk < 128; kk += 32) {
      v16bf a = frag_ld((const unsigned*)(qs_ + (wave * 16 + l16) * Dn + d0 + kk), half);
#pragma unroll
      for (int f = 0; f < 4; ++f) {
        v16bf bm = frag_ld((const unsigned*)(kcur + (f * 16 + l16) * 128 + kk), half);
        sc[f] = wmma_bf16(a, bm, sc[f]);
      }
    }
  }
}

// ---------------------------------------------------------------------------
// Kernel 2: online-softmax statistics (row max m, row sumexp l).
//   Block = 128 threads (4 waves) = 64 query rows of one batch.
// ---------------------------------------------------------------------------
__global__ __launch_bounds__(128)
void attn_stats_kernel(const bf16* __restrict__ qb, const bf16* __restrict__ kb,
                       float* __restrict__ mbuf, float* __restrict__ lbuf) {
  extern __shared__ char smem[];
  const bf16* qs_ = (const bf16*)smem;
  const bf16* kc_ = (const bf16*)(smem + KC_OFF);

  const int tid  = threadIdx.x;
  const int lane = tid & 31;
  const int wave = tid >> 5;
  const int half = lane >> 4;
  const int l16  = lane & 15;
  const int qt   = blockIdx.x;
  const int b    = blockIdx.y;

  if (wave == 0)   // q tile: one TDM descriptor, stays resident all kt
    tdm_load_2d(QS_OFF, qb + ((size_t)b * Sn + qt * 64) * Dn, 1024, 64, 1024);

  float m[8], l[8];
#pragma unroll
  for (int r = 0; r < 8; ++r) { m[r] = -1e30f; l[r] = 0.0f; }

  for (int kt = 0; kt <= qt; ++kt) {
    const bf16* ksrc = kb + ((size_t)b * Sn + kt * 64) * Dn;
    if (wave == 0) tdm_load_2d(KC_OFF, ksrc, 128, 64, Dn);   // chunk 0
    v8f sc[4] = {};
    score_tile(qs_, kc_, ksrc, wave, l16, half, sc);

    // online softmax update for this lane's 8 rows (rows span 16 lanes)
#pragma unroll
    for (int r = 0; r < 8; ++r) {
      int qrow = qt * 64 + wave * 16 + half * 8 + r;
      float vals[4];
      float rmax = -1e30f;
#pragma unroll
      for (int f = 0; f < 4; ++f) {
        float s = sc[f][r] * SCALE;
        int key = kt * 64 + f * 16 + l16;
        if (key > qrow) s = -1e30f;     // causal (only bites when kt==qt)
        vals[f] = s;
        rmax = fmaxf(rmax, s);
      }
#pragma unroll
      for (int sh = 1; sh < 16; sh <<= 1)
        rmax = fmaxf(rmax, __shfl_xor(rmax, sh, 32));
      float mnew = fmaxf(m[r], rmax);
      float rsum = 0.0f;
#pragma unroll
      for (int f = 0; f < 4; ++f) rsum += __expf(vals[f] - mnew);
#pragma unroll
      for (int sh = 1; sh < 16; sh <<= 1) rsum += __shfl_xor(rsum, sh, 32);
      l[r] = l[r] * __expf(m[r] - mnew) + rsum;
      m[r] = mnew;
    }
  }

  if (l16 == 0) {    // lanes 0 and 16 each own 8 rows
#pragma unroll
    for (int r = 0; r < 8; ++r) {
      size_t idx = (size_t)b * Sn + qt * 64 + wave * 16 + half * 8 + r;
      mbuf[idx] = m[r];
      lbuf[idx] = l[r];
    }
  }
}

// ---------------------------------------------------------------------------
// Kernel 3: materialize probabilities  P = exp(scale*qk^T - m) / l  (bf16).
//   One block per (kt, qt, b) 64x64 tile; upper-triangular tiles exit early.
// ---------------------------------------------------------------------------
__global__ __launch_bounds__(128)
void attn_probs_kernel(const bf16* __restrict__ qb, const bf16* __restrict__ kb,
                       const float* __restrict__ mbuf, const float* __restrict__ lbuf,
                       bf16* __restrict__ Pbuf) {
  const int kt = blockIdx.x;
  const int qt = blockIdx.y;
  const int b  = blockIdx.z;
  if (kt > qt) return;

  extern __shared__ char smem[];
  const bf16* qs_ = (const bf16*)smem;
  const bf16* kc_ = (const bf16*)(smem + KC_OFF);

  const int tid  = threadIdx.x;
  const int lane = tid & 31;
  const int wave = tid >> 5;
  const int half = lane >> 4;
  const int l16  = lane & 15;

  const bf16* ksrc = kb + ((size_t)b * Sn + kt * 64) * Dn;
  if (wave == 0) {
    tdm_load_2d(QS_OFF, qb + ((size_t)b * Sn + qt * 64) * Dn, 1024, 64, 1024);
    tdm_load_2d(KC_OFF, ksrc, 128, 64, Dn);
  }

  float mrow[8], linv[8];
#pragma unroll
  for (int r = 0; r < 8; ++r) {
    size_t idx = (size_t)b * Sn + qt * 64 + wave * 16 + half * 8 + r;
    mrow[r] = mbuf[idx];
    linv[r] = 1.0f / lbuf[idx];
  }

  v8f sc[4] = {};
  score_tile(qs_, kc_, ksrc, wave, l16, half, sc);

#pragma unroll
  for (int f = 0; f < 4; ++f) {
    int key = kt * 64 + f * 16 + l16;
#pragma unroll
    for (int r = 0; r < 8; ++r) {
      int qrow = qt * 64 + wave * 16 + half * 8 + r;
      float p = __expf(sc[f][r] * SCALE - mrow[r]) * linv[r];
      if (key > qrow) p = 0.0f;
      Pbuf[((size_t)b * Sn + qrow) * Sn + key] = (bf16)p;
    }
  }
}

// ---------------------------------------------------------------------------
// Kernel 4: O = P @ V (causal: skip tiles above the diagonal).
//   Block = (dch, qt, b): 64 q rows x 128 d cols, 4 waves, f32 output.
//   P tile staged via async global->LDS (ASYNCcnt); V transposed via packed
//   b32 LDS stores.  Dynamic LDS: ps_ at 0 (8 KB), vt_ at 8 KB (16 KB).
// ---------------------------------------------------------------------------
constexpr unsigned PS_OFF = 0;
constexpr unsigned VT_OFF = 64 * 64 * 2;                 // 8192
constexpr int SMEM_PV = (int)(VT_OFF + 128 * 64 * 2);    // 24 KB

__global__ __launch_bounds__(128)
void attn_pv_kernel(const bf16* __restrict__ Pbuf, const bf16* __restrict__ vb,
                    float* __restrict__ out) {
  extern __shared__ char smem[];
  const bf16* ps_ = (const bf16*)(smem + PS_OFF);   // [qrow][key] 64x64
  bf16*       vt_ = (bf16*)(smem + VT_OFF);         // [d][key]   128x64

  const int tid  = threadIdx.x;
  const int lane = tid & 31;
  const int wave = tid >> 5;
  const int half = lane >> 4;
  const int l16  = lane & 15;
  const int dch  = blockIdx.x;
  const int qt   = blockIdx.y;
  const int b    = blockIdx.z;

  v8f oacc[8] = {};

  for (int kt = 0; kt <= qt; ++kt) {
    __syncthreads();   // prior iteration finished reading ps_/vt_
    // --- P tile via async loads: 64 rows x 128 B, row stride Sn*2 B ---
    {
      const bf16* psrc = Pbuf + ((size_t)b * Sn + qt * 64) * Sn + kt * 64;
      for (int i = tid; i < 512; i += 128) {         // 512 x 16 B
        int row = i >> 3, c = i & 7;
        async_ld_b128(PS_OFF + (unsigned)i * 16, psrc,
                      (int)(((size_t)row * Sn + c * 8) * sizeof(bf16)));
      }
    }
    // --- V chunk transposed, key-pair packed (b32 LDS stores) ---
    {
      int kp   = tid & 31;            // keys 2kp, 2kp+1
      int dseg = (tid >> 5) * 32;     // 32 d values per thread
      const bf16* r0 = vb + ((size_t)b * Sn + kt * 64 + 2 * kp) * Dn + dch * 128 + dseg;
      const bf16* r1 = r0 + Dn;
      uint4 ta[4], tb[4];
#pragma unroll
      for (int i = 0; i < 4; ++i) { ta[i] = ((const uint4*)r0)[i]; tb[i] = ((const uint4*)r1)[i]; }
      const bf16* fa = (const bf16*)ta;
      const bf16* fb = (const bf16*)tb;
#pragma unroll
      for (int j = 0; j < 32; ++j) {
        union { bf16 h[2]; unsigned u; } pk;
        pk.h[0] = fa[j]; pk.h[1] = fb[j];
        *(unsigned*)&vt_[(dseg + j) * 64 + 2 * kp] = pk.u;
      }
    }
    wait_async0();       // ASYNCcnt -> 0: P tile landed in LDS
    __syncthreads();
#pragma unroll
    for (int kk = 0; kk < 64; kk += 32) {
      v16bf a = frag_ld((const unsigned*)(ps_ + (wave * 16 + l16) * 64 + kk), half);
#pragma unroll
      for (int f = 0; f < 8; ++f) {
        v16bf bm = frag_ld((const unsigned*)(vt_ + (f * 16 + l16) * 64 + kk), half);
        oacc[f] = wmma_bf16(a, bm, oacc[f]);
      }
    }
  }

#pragma unroll
  for (int f = 0; f < 8; ++f) {
    int col = dch * 128 + f * 16 + l16;
#pragma unroll
    for (int r = 0; r < 8; ++r) {
      size_t row = (size_t)b * Sn + qt * 64 + wave * 16 + half * 8 + r;
      out[row * Dn + col] = oacc[f][r];
    }
  }
}

// ---------------------------------------------------------------------------
extern "C" void kernel_launch(void* const* d_in, const int* in_sizes, int n_in,
                              void* d_out, int out_size, void* d_ws, size_t ws_size,
                              hipStream_t stream) {
  (void)in_sizes; (void)n_in; (void)out_size; (void)ws_size;
  const float* Q  = (const float*)d_in[0];
  const float* K  = (const float*)d_in[1];
  const float* V  = (const float*)d_in[2];
  const float* Wq = (const float*)d_in[3];
  const float* Wk = (const float*)d_in[4];
  const float* Wv = (const float*)d_in[5];
  const float* bq = (const float*)d_in[6];
  const float* bk = (const float*)d_in[7];
  const float* bv = (const float*)d_in[8];
  // d_in[9] = causal mask -- implemented analytically.
  float* out = (float*)d_out;

  char* ws = (char*)d_ws;
  const size_t qkvBytes   = (size_t)Bn * Sn * Dn * sizeof(bf16);   // 32 MB each
  const size_t statsBytes = (size_t)Bn * Sn * sizeof(float);       // 64 KB each
  bf16*  qb   = (bf16*)(ws);
  bf16*  kb   = (bf16*)(ws + qkvBytes);
  bf16*  vb   = (bf16*)(ws + 2 * qkvBytes);
  float* mbuf = (float*)(ws + 3 * qkvBytes);
  float* lbuf = (float*)(ws + 3 * qkvBytes + statsBytes);
  bf16*  Pbuf = (bf16*)(ws + 3 * qkvBytes + 2 * statsBytes);       // 64 MB

  dim3 pgrid(Dn / 128, (Bn * Sn) / 128);
  proj_kernel<<<pgrid, 256, 0, stream>>>(Q, Wq, bq, qb);
  proj_kernel<<<pgrid, 256, 0, stream>>>(K, Wk, bk, kb);
  proj_kernel<<<pgrid, 256, 0, stream>>>(V, Wv, bv, vb);

  (void)hipFuncSetAttribute((const void*)attn_stats_kernel,
                            hipFuncAttributeMaxDynamicSharedMemorySize, SMEM_SCORE);
  (void)hipFuncSetAttribute((const void*)attn_probs_kernel,
                            hipFuncAttributeMaxDynamicSharedMemorySize, SMEM_SCORE);

  attn_stats_kernel<<<dim3(Sn / 64, Bn), 128, SMEM_SCORE, stream>>>(qb, kb, mbuf, lbuf);
  attn_probs_kernel<<<dim3(Sn / 64, Sn / 64, Bn), 128, SMEM_SCORE, stream>>>(qb, kb, mbuf, lbuf, Pbuf);
  attn_pv_kernel<<<dim3(Dn / 128, Sn / 64, Bn), 128, SMEM_PV, stream>>>(Pbuf, vb, out);
}
